// OuterProductMean_55688545960637
// MI455X (gfx1250) — compile-verified, matrix operand order checked by reading, not run on hardware
//
#include <hip/hip_runtime.h>

typedef __attribute__((ext_vector_type(16))) __bf16 v16bf;
typedef __attribute__((ext_vector_type(8)))  float  v8f;

#define C_M   256
#define C_H   32
#define C_Z   128
#define NSEQ  128
#define S_DIM 256
#define LDSS  1032   // 1024 + 8 bf16 pad -> row stride 2064B, rotates banks

// Load a 16x32 bf16 WMMA A/B fragment for one lane.
// rowBase points at this lane's (M or N) row, K contiguous in memory.
// Layout per ISA: lane<16 holds K = k0+0..7 and k0+16..23; lane>=16 holds +8.
static __device__ __forceinline__ v16bf load_frag16(const __bf16* rowBase, int k0, int lane) {
    union { v16bf v; uint4 q[2]; } f;
    const uint4* p = reinterpret_cast<const uint4*>(rowBase + k0 + ((lane >> 4) << 3));
    f.q[0] = p[0];   // 8 bf16: K = k0(+8)..+7
    f.q[1] = p[2];   // 8 bf16: K = k0(+8)+16..+23
    return f.v;
}

// ---------------- Prep 1: LayerNorm + dual projection, bf16 [pos][c][s] ----
__global__ __launch_bounds__(256) void ln_proj_kernel(
    const float* __restrict__ m, const float* __restrict__ ln_w, const float* __restrict__ ln_b,
    const float* __restrict__ Wa, const float* __restrict__ ba,
    const float* __restrict__ Wb, const float* __restrict__ bb,
    __bf16* __restrict__ a_ws, __bf16* __restrict__ b_ws) {
    __shared__ float mnLDS[8][C_M];
    const int lane = threadIdx.x & 31;
    const int wave = threadIdx.x >> 5;
    const int rid  = blockIdx.x * 8 + wave;   // 0..32767 = (s, i)
    const int s    = rid >> 8;                // 0..127
    const int i    = rid & 255;               // 0..255
    const float* row = m + (size_t)rid * C_M;

    float x[8], sum = 0.f, sq = 0.f;
    #pragma unroll
    for (int t = 0; t < 8; ++t) { x[t] = row[lane * 8 + t]; sum += x[t]; sq += x[t] * x[t]; }
    #pragma unroll
    for (int o = 16; o > 0; o >>= 1) { sum += __shfl_xor(sum, o, 32); sq += __shfl_xor(sq, o, 32); }
    const float mu   = sum * (1.f / C_M);
    const float var  = sq * (1.f / C_M) - mu * mu;
    const float rstd = rsqrtf(var + 1e-5f);
    #pragma unroll
    for (int t = 0; t < 8; ++t) {
        const int k = lane * 8 + t;
        mnLDS[wave][k] = (x[t] - mu) * rstd * ln_w[k] + ln_b[k];
    }
    __syncthreads();
    const int c = lane;                       // c_h = 32 = one lane per channel
    float accA = ba[c], accB = bb[c];
    for (int k = 0; k < C_M; ++k) {
        const float v = mnLDS[wave][k];
        accA = fmaf(v, Wa[k * C_H + c], accA);
        accB = fmaf(v, Wb[k * C_H + c], accB);
    }
    a_ws[((size_t)i * C_H + c) * NSEQ + s] = (__bf16)accA;
    b_ws[((size_t)i * C_H + c) * NSEQ + s] = (__bf16)accB;
}

// ---------------- Prep 2: Wo[cd][cz] f32 -> WoT[cz][cd] bf16 ---------------
__global__ __launch_bounds__(256) void wo_prep_kernel(const float* __restrict__ Wo,
                                                      __bf16* __restrict__ woT) {
    const int t = blockIdx.x * 256 + threadIdx.x;  // 0..131071
    const int z = t >> 10, k = t & 1023;
    woT[t] = (__bf16)Wo[(size_t)k * C_Z + z];
}

// ---------------- Main: fused outer-product + Wo projection ----------------
// block = 128 threads (4 waves); grid = (16 j-blocks, 256 i)
__global__ __launch_bounds__(128) void opm_kernel(
    const __bf16* __restrict__ a_ws, const __bf16* __restrict__ b_ws,
    const __bf16* __restrict__ woT, const float* __restrict__ bo,
    float* __restrict__ out) {
    __shared__ __bf16 outerLDS[16 * LDSS];
    const int lane = threadIdx.x & 31;
    const int wave = threadIdx.x >> 5;
    const int l15  = lane & 15;
    const int rofs = (lane >> 4) * 8;
    const int i    = blockIdx.y;
    const int j0   = blockIdx.x * 16;

    // ---- Stage 1: outer[c,d] = sum_s a[s,i,c] * b[s,j,d], 4 pairs/wave ----
    const __bf16* aRow0 = a_ws + (size_t)i * C_H * NSEQ;
    #pragma unroll 1
    for (int pp = 0; pp < 4; ++pp) {
        const int p = wave * 4 + pp;
        const int j = j0 + p;
        const __bf16* bRow0 = b_ws + (size_t)j * C_H * NSEQ;
        v8f acc00 = {}, acc01 = {}, acc10 = {}, acc11 = {};
        #pragma unroll
        for (int ks = 0; ks < NSEQ; ks += 32) {
            v16bf a0 = load_frag16(aRow0 + (size_t)(l15)      * NSEQ, ks, lane);
            v16bf a1 = load_frag16(aRow0 + (size_t)(16 + l15) * NSEQ, ks, lane);
            v16bf b0 = load_frag16(bRow0 + (size_t)(l15)      * NSEQ, ks, lane);
            v16bf b1 = load_frag16(bRow0 + (size_t)(16 + l15) * NSEQ, ks, lane);
            acc00 = __builtin_amdgcn_wmma_f32_16x16x32_bf16(false, a0, false, b0, (short)0, acc00, false, false);
            acc01 = __builtin_amdgcn_wmma_f32_16x16x32_bf16(false, a0, false, b1, (short)0, acc01, false, false);
            acc10 = __builtin_amdgcn_wmma_f32_16x16x32_bf16(false, a1, false, b0, (short)0, acc10, false, false);
            acc11 = __builtin_amdgcn_wmma_f32_16x16x32_bf16(false, a1, false, b1, (short)0, acc11, false, false);
        }
        const float inv = 1.f / (float)NSEQ;
        #pragma unroll
        for (int r = 0; r < 8; ++r) {
            const int c0 = r + rofs;      // row within 16x16 tile (C/D layout)
            outerLDS[p * LDSS + (c0)      * C_H + (l15)]      = (__bf16)(acc00[r] * inv);
            outerLDS[p * LDSS + (c0)      * C_H + (16 + l15)] = (__bf16)(acc01[r] * inv);
            outerLDS[p * LDSS + (16 + c0) * C_H + (l15)]      = (__bf16)(acc10[r] * inv);
            outerLDS[p * LDSS + (16 + c0) * C_H + (16 + l15)] = (__bf16)(acc11[r] * inv);
        }
    }
    __syncthreads();

    // ---- Stage 2: z[p, cz] = outer[p, cd] @ WoT[cz, cd]; 2 N-tiles/wave ----
    // unroll 4 so the compiler can clause 4 LDS A-loads + 8 global B-loads
    // and hide them behind 8 WMMAs (same pipelining it achieved in stage 1).
    v8f acc2[2] = {{}, {}};
    #pragma unroll 4
    for (int kk = 0; kk < C_H * C_H; kk += 32) {
        v16bf a2 = load_frag16(&outerLDS[(size_t)l15 * LDSS], kk, lane);
        #pragma unroll
        for (int t = 0; t < 2; ++t) {
            const int cz0 = (wave * 2 + t) * 16;
            v16bf b2 = load_frag16(woT + (size_t)(cz0 + l15) * (C_H * C_H), kk, lane);
            acc2[t] = __builtin_amdgcn_wmma_f32_16x16x32_bf16(false, a2, false, b2, (short)0, acc2[t], false, false);
        }
    }
    #pragma unroll
    for (int t = 0; t < 2; ++t) {
        const int cz = (wave * 2 + t) * 16 + l15;
        const float bias = bo[cz];
        #pragma unroll
        for (int r = 0; r < 8; ++r) {
            const int j = j0 + r + rofs;
            out[(((size_t)i * S_DIM) + j) * C_Z + cz] = acc2[t][r] + bias;
        }
    }
}

extern "C" void kernel_launch(void* const* d_in, const int* in_sizes, int n_in,
                              void* d_out, int out_size, void* d_ws, size_t ws_size,
                              hipStream_t stream) {
    const float* m    = (const float*)d_in[0];
    const float* ln_w = (const float*)d_in[1];
    const float* ln_b = (const float*)d_in[2];
    const float* Wa   = (const float*)d_in[3];
    const float* ba   = (const float*)d_in[4];
    const float* Wb   = (const float*)d_in[5];
    const float* bb   = (const float*)d_in[6];
    const float* Wo   = (const float*)d_in[7];
    const float* bo   = (const float*)d_in[8];
    float* out = (float*)d_out;

    __bf16* a_ws = (__bf16*)d_ws;                          // 256*32*128 bf16 = 2 MB
    __bf16* b_ws = a_ws + (size_t)S_DIM * C_H * NSEQ;      // 2 MB
    __bf16* woT  = b_ws + (size_t)S_DIM * C_H * NSEQ;      // 128*1024 bf16 = 256 KB

    ln_proj_kernel<<<4096, 256, 0, stream>>>(m, ln_w, ln_b, Wa, ba, Wb, bb, a_ws, b_ws);
    wo_prep_kernel<<<512, 256, 0, stream>>>(Wo, woT);
    opm_kernel<<<dim3(16, 256), 128, 0, stream>>>(a_ws, b_ws, woT, bo, out);
}
